// RRID_44805098831987
// MI455X (gfx1250) — matrix operand order repaired and checked by reading, not compiled, not run on hardware
//
#include <hip/hip_runtime.h>
#include <math.h>

// ---------------------------------------------------------------------------
// CDNA5 (gfx1250) WMMA / TDM types
// ---------------------------------------------------------------------------
typedef __attribute__((ext_vector_type(16))) __bf16 v16bf;
typedef __attribute__((ext_vector_type(8)))  float  v8f;
typedef __attribute__((ext_vector_type(4)))  unsigned int v4u;
typedef __attribute__((ext_vector_type(8)))  int    v8i;
typedef __attribute__((ext_vector_type(4)))  int    v4i;

__device__ __forceinline__ __bf16 f2bf(float f) {
    unsigned u = __builtin_bit_cast(unsigned, f);
    unsigned short r = (unsigned short)((u + 0x7FFFu + ((u >> 16) & 1u)) >> 16);
    return __builtin_bit_cast(__bf16, r);
}

struct alignas(32) Frag32 { uint4 lo; uint4 hi; };

// A-fragment (16x32 bf16, MxK): lane row = lane&15, half h = lane>>4.
// VGPRs 0-3 hold K = k0 + h*8 .. +7 ; VGPRs 4-7 hold K = k0+16 + h*8 .. +7
__device__ __forceinline__ v16bf load_frag_a(const __bf16* p) {
    Frag32 t;
    t.lo = *(const uint4*)(p);        // 8 bf16
    t.hi = *(const uint4*)(p + 16);   // 8 bf16, 32B further
    return __builtin_bit_cast(v16bf, t);
}
// B-fragment (32x16 bf16, KxN): lane col = lane&15, half h = lane>>4.
// VGPRs 0-3 hold K = k0 + h*8 .. +7 ; VGPRs 4-7 hold K = k0+16 + h*8 .. +7
__device__ __forceinline__ v16bf load_frag_b(const __bf16* p) {
    Frag32 t;
    t.lo = *(const uint4*)(p);
    t.hi = *(const uint4*)(p + 16);
    return __builtin_bit_cast(v16bf, t);
}

// ---------------------------------------------------------------------------
// Implicit-GEMM conv (TAPS=9: 3x3 pad=1, TAPS=1: 1x1), C=64 -> K=64,
// H=W=256 fixed, via v_wmma_f32_16x16x32_bf16.
// GEMM K-order: k = c*TAPS + (ky*3+kx), so the OIHW weight blob flattened per
// output channel IS the column-major B matrix.
//
// Tile: 64 output pixels x 64 output channels per workgroup, 4 waves.
//   wave w owns M-tile w (16 pixels) and all four 16-wide N tiles (4 accums).
// B matrix (64 x Ktot bf16) is staged global->LDS with one Tensor-Data-Mover
// descriptor (tensor_load_to_lds, TENSORcnt). The A patch matrix is built in
// LDS from one coalesced read per raw input element (each value feeds up to
// 3 kx taps).
// ACT: 0 none, 1 gelu(exact), 2 relu, 3 sigmoid
// EPI: 0 none, 1 add extra[], 2 mul extra[]  (applied after activation)
// ---------------------------------------------------------------------------
template <int TAPS, int ACT, int EPI>
__launch_bounds__(128)
__global__ void conv_wmma_kernel(const float* __restrict__ in,
                                 const __bf16* __restrict__ wB,
                                 const float* __restrict__ bias,
                                 const float* __restrict__ extra,
                                 float* __restrict__ out)
{
    constexpr int C    = 64;
    constexpr int H    = 256;
    constexpr int W    = 256;
    constexpr int Ktot = TAPS * C;            // 576 (3x3) or 64 (1x1)
    constexpr int TP   = 64;                  // pixels per workgroup

    extern __shared__ char smem[];
    __bf16* ldsA = (__bf16*)smem;                       // TP x Ktot
    __bf16* ldsB = (__bf16*)(smem + TP * Ktot * 2);     // 64 x Ktot

    const int pix_base = blockIdx.x * TP;     // 64 consecutive x in one row
    const int n_img = pix_base >> 16;         // / (256*256)
    const int y  = (pix_base >> 8) & 255;
    const int xb = pix_base & 255;            // multiple of 64

    const int tid = threadIdx.x;

    // ---- stage B (weights) into LDS with the Tensor Data Mover (wave 0) ----
#if __has_builtin(__builtin_amdgcn_tensor_load_to_lds)
    if (tid < 32) {
        unsigned long long ga = (unsigned long long)(uintptr_t)wB;
        unsigned lds_off = (unsigned)(uintptr_t)ldsB;   // low 32 bits = LDS addr
        constexpr unsigned wsize = 64u * (unsigned)Ktot; // elements (2B each)
        v4u g0 = { 1u,                                   // count=1, user mode
                   lds_off,
                   (unsigned)(ga & 0xFFFFFFFFu),
                   (unsigned)((ga >> 32) & 0x1FFFFFFu) | (2u << 30) }; // type=2
        v8i g1 = { (int)(1u << 16),                      // data_size=1 (2 bytes)
                   (int)((wsize & 0xFFFFu) << 16),       // tensor_dim0[15:0]
                   (int)(((wsize >> 16) & 0xFFFFu) | (1u << 16)), // dim0 hi | tensor_dim1=1
                   (int)((wsize & 0xFFFFu) << 16),       // tile_dim0 = wsize
                   (int)1,                               // tile_dim1=1, tile_dim2=0
                   (int)wsize,                           // tensor_dim0_stride lo32
                   0, 0 };
        v4i gz4 = { 0, 0, 0, 0 };
        v8i gz8 = { 0, 0, 0, 0, 0, 0, 0, 0 };
        __builtin_amdgcn_tensor_load_to_lds(g0, g1, gz4, gz4, gz8, 0);
    }
#else
    for (int idx = tid; idx < 64 * Ktot; idx += 128) ldsB[idx] = wB[idx];
#endif

    // ---- build A patch matrix [TP pixels][Ktot] in LDS (bf16, row-major) ----
    if (TAPS == 9) {
        // one coalesced global read per raw element (c, ky, ox), scattered to
        // up to 3 kx taps:  A[m][c*9 + ky*3 + kx] = in[c][y+ky-1][xb+m+kx-1],
        // i.e. m = ox - kx for ox = ix - xb + 1 in [0, 66).
        constexpr int FILL = 64 * 3 * 66;     // 12672
        for (int idx = tid; idx < FILL; idx += 128) {
            int c  = idx / 198;
            int r  = idx - c * 198;
            int ky = r / 66;
            int ox = r - ky * 66;
            int iy = y + ky - 1;
            int ix = xb + ox - 1;
            float v = 0.0f;
            if (iy >= 0 && iy < H && ix >= 0 && ix < W)
                v = in[((n_img * C + c) * H + iy) * W + ix];
            __bf16 bv = f2bf(v);
            __bf16* dst = ldsA + c * 9 + ky * 3;
#pragma unroll
            for (int kx = 0; kx < 3; ++kx) {
                int m = ox - kx;
                if (m >= 0 && m < TP) dst[m * Ktot + kx] = bv;
            }
        }
    } else {
        // 1x1: A[m][c] = in[c][y][xb+m]; coalesced over m
        constexpr int FILL = TP * 64;         // 4096
        for (int idx = tid; idx < FILL; idx += 128) {
            int c = idx >> 6;
            int m = idx & 63;
            float v = in[((n_img * C + c) * H + y) * W + xb + m];
            ldsA[m * Ktot + c] = f2bf(v);
        }
    }
#if __has_builtin(__builtin_amdgcn_s_wait_tensorcnt)
    if (tid < 32) __builtin_amdgcn_s_wait_tensorcnt(0);
#endif
    __syncthreads();

    // ---- WMMA main loop: each wave = 16 pixels x 64 channels ----
    const int lane   = tid & 31;
    const int m_tile = tid >> 5;              // wave id 0..3
    const int col    = lane & 15;
    const int hh     = lane >> 4;

    const __bf16* pa = ldsA + (m_tile * 16 + col) * Ktot + hh * 8;
    const __bf16* pb = ldsB + col * Ktot + hh * 8;

    v8f acc[4] = { v8f{}, v8f{}, v8f{}, v8f{} };
    constexpr int kchunks = Ktot >> 5;        // 18 or 2
    for (int kc = 0; kc < kchunks; ++kc) {
        const int k0 = kc * 32;
        v16bf a = load_frag_a(pa + k0);
#pragma unroll
        for (int nt = 0; nt < 4; ++nt) {
            v16bf b = load_frag_b(pb + nt * 16 * Ktot + k0);
            acc[nt] = __builtin_amdgcn_wmma_f32_16x16x32_bf16(
                          false, a, false, b, (short)0, acc[nt], false, false);
        }
    }

    // ---- epilogue: C/D layout => lane col = lane&15, rows = hh*8 + r ----
    const int xo = xb + m_tile * 16 + hh * 8;     // 8-aligned -> 32B aligned
#pragma unroll
    for (int nt = 0; nt < 4; ++nt) {
        const int chan = nt * 16 + col;
        const int base = ((n_img * C + chan) * H + y) * W + xo;
        const float bch = bias[chan];
        float v[8];
#pragma unroll
        for (int r = 0; r < 8; ++r) {
            float t = acc[nt][r] + bch;
            if constexpr (ACT == 1) t = 0.5f * t * (1.0f + erff(t * 0.70710678118f));
            else if constexpr (ACT == 2) t = fmaxf(t, 0.0f);
            else if constexpr (ACT == 3) t = 1.0f / (1.0f + expf(-t));
            v[r] = t;
        }
        if constexpr (EPI != 0) {
            float4 e0 = *(const float4*)(extra + base);
            float4 e1 = *(const float4*)(extra + base + 4);
            if constexpr (EPI == 1) {
                v[0] += e0.x; v[1] += e0.y; v[2] += e0.z; v[3] += e0.w;
                v[4] += e1.x; v[5] += e1.y; v[6] += e1.z; v[7] += e1.w;
            } else {
                v[0] *= e0.x; v[1] *= e0.y; v[2] *= e0.z; v[3] *= e0.w;
                v[4] *= e1.x; v[5] *= e1.y; v[6] *= e1.z; v[7] *= e1.w;
            }
        }
        float4 o0 = { v[0], v[1], v[2], v[3] };
        float4 o1 = { v[4], v[5], v[6], v[7] };
        *(float4*)(out + base)     = o0;
        *(float4*)(out + base + 4) = o1;
    }
}

// ---------------------------------------------------------------------------
// f32 -> bf16 weight pack (OIHW flat blob == column-major GEMM B matrix)
// ---------------------------------------------------------------------------
__global__ void pack_bf16_kernel(const float* __restrict__ src,
                                 __bf16* __restrict__ dst, int n) {
    int i = blockIdx.x * blockDim.x + threadIdx.x;
    if (i < n) dst[i] = f2bf(src[i]);
}

// ---------------------------------------------------------------------------
// Depthwise conv 3x3, stride 8, pad 2, dilation 2: (8,64,256,256)->(8,64,32,32)
// ---------------------------------------------------------------------------
__global__ void dwconv_kernel(const float* __restrict__ in,
                              const float* __restrict__ w,
                              const float* __restrict__ b,
                              float* __restrict__ out) {
    int i = blockIdx.x * blockDim.x + threadIdx.x;   // 524288 total
    int ox = i & 31; int t = i >> 5;
    int oy = t & 31; t >>= 5;
    int c  = t & 63; int n = t >> 6;
    float acc = b[c];
    const float* plane = in + (size_t)(n * 64 + c) * 256 * 256;
#pragma unroll
    for (int ky = 0; ky < 3; ++ky) {
        int iy = oy * 8 + ky * 2 - 2;
        if (iy < 0 || iy >= 256) continue;
#pragma unroll
        for (int kx = 0; kx < 3; ++kx) {
            int ix = ox * 8 + kx * 2 - 2;
            if (ix < 0 || ix >= 256) continue;
            acc += plane[iy * 256 + ix] * w[c * 9 + ky * 3 + kx];
        }
    }
    out[i] = acc;
}

// ---------------------------------------------------------------------------
// 8x8 block iDCT: out[j1][j2] = sum_k1,k2 T[j1][k1]*T[j2][k2]*in[k1][k2]
// with T[j][k] = cos(pi/4 * j*k)/sqrt(8)  (real(ifft ortho) twice)
// ---------------------------------------------------------------------------
__global__ void iddct_kernel(const float* __restrict__ in, float* __restrict__ out) {
    __shared__ float T[64];
    int tid = threadIdx.x;
    if (tid < 64) {
        int j = tid >> 3, k = tid & 7;
        T[tid] = cosf(0.7853981633974483f * (float)(j * k)) * 0.3535533905932738f;
    }
    __syncthreads();
    int i = blockIdx.x * blockDim.x + tid;           // 524288 total
    int ox = i & 31; int t = i >> 5;
    int oy = t & 31; int plane = t >> 5;             // 512 planes of 32x32
    int bx = ox >> 3, j2 = ox & 7;
    int by = oy >> 3, j1 = oy & 7;
    const float* p = in + plane * 1024;
    float acc = 0.0f;
#pragma unroll
    for (int k1 = 0; k1 < 8; ++k1) {
        float s = 0.0f;
#pragma unroll
        for (int k2 = 0; k2 < 8; ++k2)
            s += T[j2 * 8 + k2] * p[(by * 8 + k1) * 32 + bx * 8 + k2];
        acc += T[j1 * 8 + k1] * s;
    }
    out[i] = acc;
}

// ---------------------------------------------------------------------------
// Bilinear upsample 32x32 -> 256x256, half-pixel centers, edge clamp
// ---------------------------------------------------------------------------
__global__ void upsample_kernel(const float* __restrict__ in, float* __restrict__ out) {
    int i = blockIdx.x * blockDim.x + threadIdx.x;   // 33554432 total
    int ox = i & 255; int t = i >> 8;
    int oy = t & 255; int plane = t >> 8;            // 512 planes
    float sy = (oy + 0.5f) * 0.125f - 0.5f;
    float sx = (ox + 0.5f) * 0.125f - 0.5f;
    float fy0 = floorf(sy), fx0 = floorf(sx);
    int y0 = (int)fy0, x0 = (int)fx0;
    float wy = sy - fy0, wx = sx - fx0;
    int y0c = min(max(y0, 0), 31), y1c = min(max(y0 + 1, 0), 31);
    int x0c = min(max(x0, 0), 31), x1c = min(max(x0 + 1, 0), 31);
    const float* p = in + plane * 1024;
    float v00 = p[y0c * 32 + x0c], v01 = p[y0c * 32 + x1c];
    float v10 = p[y1c * 32 + x0c], v11 = p[y1c * 32 + x1c];
    float v0 = v00 + (v01 - v00) * wx;
    float v1 = v10 + (v11 - v10) * wx;
    out[i] = v0 + (v1 - v0) * wy;
}

// ---------------------------------------------------------------------------
// Host launcher
// ---------------------------------------------------------------------------
extern "C" void kernel_launch(void* const* d_in, const int* in_sizes, int n_in,
                              void* d_out, int out_size, void* d_ws, size_t ws_size,
                              hipStream_t stream) {
    (void)in_sizes; (void)n_in; (void)out_size; (void)ws_size;

    const float* x       = (const float*)d_in[0];
    const float* conv_w  = (const float*)d_in[1];
    const float* conv_b  = (const float*)d_in[2];
    const float* ddct_w  = (const float*)d_in[3];
    const float* ddct_b  = (const float*)d_in[4];
    const float* dct_w   = (const float*)d_in[5];
    const float* dct_b   = (const float*)d_in[6];
    const float* w1_w    = (const float*)d_in[7];
    const float* w1_b    = (const float*)d_in[8];
    const float* w3_w    = (const float*)d_in[9];
    const float* w3_b    = (const float*)d_in[10];
    const float* after_w = (const float*)d_in[11];
    const float* after_b = (const float*)d_in[12];

    const int N = 8, C = 64, H = 256, W = 256;
    const size_t BIG   = (size_t)N * C * H * W * sizeof(float);   // 134 MB
    const size_t SMALL = (size_t)N * C * 32 * 32 * sizeof(float); // 2 MB

    char* ws = (char*)d_ws;
    float*  buf1   = (float*)(ws);                                // dct_feat / relu feat
    float*  buf2   = (float*)(ws + BIG);                          // out / up / prod
    float*  small0 = (float*)(ws + 2 * BIG);                      // dct_feat_0
    float*  small1 = (float*)(ws + 2 * BIG + SMALL);              // iDCT result
    __bf16* wb_conv  = (__bf16*)(ws + 2 * BIG + 2 * SMALL);
    __bf16* wb_ddct  = wb_conv  + 64 * 576;
    __bf16* wb_w1    = wb_ddct  + 64 * 576;
    __bf16* wb_after = wb_w1    + 64 * 576;
    __bf16* wb_w3    = wb_after + 64 * 576;

    // allow >64KB dynamic LDS for the 3x3 conv kernels (no-op if unsupported)
    const int SMEM3 = 4 * 64 * 576;   // A (72KB) + B (72KB) = 147456 bytes
    const int SMEM1 = 4 * 64 * 64;    // A (8KB)  + B (8KB)  = 16384 bytes
    (void)hipFuncSetAttribute((const void*)conv_wmma_kernel<9, 1, 0>,
                              hipFuncAttributeMaxDynamicSharedMemorySize, SMEM3);
    (void)hipFuncSetAttribute((const void*)conv_wmma_kernel<9, 1, 1>,
                              hipFuncAttributeMaxDynamicSharedMemorySize, SMEM3);
    (void)hipFuncSetAttribute((const void*)conv_wmma_kernel<9, 2, 0>,
                              hipFuncAttributeMaxDynamicSharedMemorySize, SMEM3);
    (void)hipFuncSetAttribute((const void*)conv_wmma_kernel<9, 0, 0>,
                              hipFuncAttributeMaxDynamicSharedMemorySize, SMEM3);
    (void)hipFuncSetAttribute((const void*)conv_wmma_kernel<1, 3, 2>,
                              hipFuncAttributeMaxDynamicSharedMemorySize, SMEM1);

    // ---- pack weights to bf16 (column-major B matrices) ----
    pack_bf16_kernel<<<(36864 + 255) / 256, 256, 0, stream>>>(conv_w,  wb_conv,  36864);
    pack_bf16_kernel<<<(36864 + 255) / 256, 256, 0, stream>>>(ddct_w,  wb_ddct,  36864);
    pack_bf16_kernel<<<(36864 + 255) / 256, 256, 0, stream>>>(w1_w,    wb_w1,    36864);
    pack_bf16_kernel<<<(36864 + 255) / 256, 256, 0, stream>>>(after_w, wb_after, 36864);
    pack_bf16_kernel<<<(4096  + 255) / 256, 256, 0, stream>>>(w3_w,    wb_w3,    4096);

    const int convBlocks = N * H * W / 64;   // 8192

    // 1) dct_feat = gelu(conv3x3(x))
    conv_wmma_kernel<9, 1, 0><<<convBlocks, 128, SMEM3, stream>>>(
        x, wb_conv, conv_b, nullptr, buf1);
    // 2) out = gelu(conv3x3(dct_feat)) + dct_feat
    conv_wmma_kernel<9, 1, 1><<<convBlocks, 128, SMEM3, stream>>>(
        buf1, wb_ddct, ddct_b, buf1, buf2);
    // 3) depthwise dilated strided conv -> (8,64,32,32)
    dwconv_kernel<<<2048, 256, 0, stream>>>(buf2, dct_w, dct_b, small0);
    // 4) 8x8 block iDCT
    iddct_kernel<<<2048, 256, 0, stream>>>(small0, small1);
    // 5) bilinear upsample to (256,256) -> buf2 (buf2 free after step 3)
    upsample_kernel<<<131072, 256, 0, stream>>>(small1, buf2);
    // 6) relu(conv3x3(x)) -> buf1 (buf1 free after step 2)
    conv_wmma_kernel<9, 2, 0><<<convBlocks, 128, SMEM3, stream>>>(
        x, wb_w1, w1_b, nullptr, buf1);
    // 7) prod = sigmoid(conv1x1(relu_feat)) * up   (elementwise in-place on buf2)
    conv_wmma_kernel<1, 3, 2><<<convBlocks, 128, SMEM1, stream>>>(
        buf1, wb_w3, w3_b, buf2, buf2);
    // 8) final conv3x3 -> d_out
    conv_wmma_kernel<9, 0, 0><<<convBlocks, 128, SMEM3, stream>>>(
        buf2, wb_after, after_b, nullptr, (float*)d_out);
}
